// ComplexLinearAndLeakyReLU_22067541967102
// MI455X (gfx1250) — matrix operand order repaired and checked by reading, not compiled, not run on hardware
//
#include <hip/hip_runtime.h>

// ---------------------------------------------------------------------------
// ComplexLinear + VN-LeakyReLU for MI455X (gfx1250, wave32, WMMA bf16 + TDM)
//
// Pipeline:
//   wprep   : f32 weights -> bf16 Wcat[256][768] (A|Bw|Cw) and Wb[256][256]
//   prep    : basis/rotation math -> bf16 T[col][k], col = i*16384 + b*1024 + n
//   gemm1   : Y[f][col]  = Wcat(256x768)  x T^T   (wmma bf16, TDM panel loads)
//   gemm2   : D[g][col]  = Wb(256x256)    x Y     (wmma bf16, vector loads)
//   vnlrelu : vector-neuron leaky relu over the 3-component axis -> out(B,F,3,N)
// ---------------------------------------------------------------------------

typedef __bf16        v16bf __attribute__((ext_vector_type(16)));
typedef float         v8f   __attribute__((ext_vector_type(8)));
typedef unsigned int  u32x4 __attribute__((ext_vector_type(4)));
typedef int           i32x4 __attribute__((ext_vector_type(4)));
typedef int           i32x8 __attribute__((ext_vector_type(8)));
typedef float         f32x4 __attribute__((ext_vector_type(4)));

union Frag16 { u32x4 q[2]; v16bf v; };

#define EPSF 1e-6f

constexpr int BQ   = 16;
constexpr int NQ   = 1024;
constexpr int EQ   = 256;
constexpr int FQ   = 256;
constexpr int CPI  = BQ * NQ;            // 16384 columns per i-plane
constexpr int NCOL = 3 * CPI;            // 49152 total GEMM columns
constexpr int K1   = 3 * EQ;             // 768 (stacked a|b|c)
constexpr int K2   = FQ;                 // 256

#if defined(__has_builtin)
#if __has_builtin(__builtin_amdgcn_tensor_load_to_lds) && \
    __has_builtin(__builtin_amdgcn_s_wait_tensorcnt)
#define HAVE_TDM 1
#endif
#endif
#ifndef HAVE_TDM
#define HAVE_TDM 0
#endif

#if HAVE_TDM
// Issue one 2-D TDM tile load: tile_d0 (contiguous, elements) x tile_d1 lines,
// line stride stride_d0 elements, 2-byte elements, into LDS at lds_off.
// Descriptor packing per CDNA5 ISA ch. 8 (D# groups 0/1; groups 2/3 unused).
__device__ __forceinline__ void tdm_load_2d(unsigned lds_off, const void* gaddr,
                                            unsigned tile_d0, unsigned tile_d1,
                                            unsigned tensor_d0,
                                            unsigned stride_d0) {
  unsigned long long ga = (unsigned long long)(uintptr_t)gaddr;
  u32x4 g0;
  g0[0] = 1u;                                          // count=1, user D#
  g0[1] = lds_off;                                     // lds_addr (bytes)
  g0[2] = (unsigned)(ga & 0xffffffffu);                // global_addr[31:0]
  g0[3] = (unsigned)((ga >> 32) & 0x01ffffffu)         // global_addr[56:32]
          | 0x80000000u;                               // type = 2 (image)
  i32x8 g1;
  g1[0] = (int)(1u << 16);                             // data_size=1 (2B), mask=0
  g1[1] = (int)((tensor_d0 & 0xffffu) << 16);          // tensor_dim0[15:0]
  g1[2] = (int)(((tensor_d0 >> 16) & 0xffffu)          // tensor_dim0[31:16]
                | (0xffffu << 16));                    // tensor_dim1[15:0] (big)
  g1[3] = (int)(0x7fffu | (tile_d0 << 16));            // tensor_dim1 hi | tile_dim0
  g1[4] = (int)(tile_d1 & 0xffffu);                    // tile_dim1, tile_dim2=0
  g1[5] = (int)stride_d0;                              // dim0 stride [31:0]
  g1[6] = 0;                                           // stride hi, dim1 stride
  g1[7] = 0;
  i32x4 z4 = {0, 0, 0, 0};
#if defined(__clang_major__) && __clang_major__ >= 23
  i32x8 z8 = {0, 0, 0, 0, 0, 0, 0, 0};
  __builtin_amdgcn_tensor_load_to_lds(g0, g1, z4, z4, z8, 0);
#else
  __builtin_amdgcn_tensor_load_to_lds(g0, g1, z4, z4, 0);
#endif
}
#endif

// ---------------------------------------------------------------------------
// Weight conversion / stacking
// ---------------------------------------------------------------------------
__global__ __launch_bounds__(256) void wprep_kernel(
    const float* __restrict__ A, const float* __restrict__ Bw,
    const float* __restrict__ Cw, const float* __restrict__ W,
    __bf16* __restrict__ Wcat, __bf16* __restrict__ Wb) {
  int idx = blockIdx.x * 256 + threadIdx.x;
  if (idx < FQ * K1) {                       // 196608
    int f = idx / K1, k = idx % K1;
    int term = k >> 8, e = k & 255;
    const float* src = (term == 0) ? A : ((term == 1) ? Bw : Cw);
    Wcat[idx] = (__bf16)src[f * EQ + e];
  } else {
    int j = idx - FQ * K1;                   // j < 65536
    Wb[j] = (__bf16)W[j];
  }
}

// ---------------------------------------------------------------------------
// Basis construction + rotated term generation (elementwise)
// ---------------------------------------------------------------------------
__global__ __launch_bounds__(256) void prep_kernel(
    const float* __restrict__ X, const float* __restrict__ J,
    __bf16* __restrict__ T) {
  const int bn = blockIdx.x;                 // 0..16383
  const int e  = threadIdx.x;                // 0..255
  const size_t base = ((size_t)bn * EQ + e) * 3;

  float jx = J[base + 0], jy = J[base + 1], jz = J[base + 2];
  float x0 = X[base + 0], x1 = X[base + 1], x2 = X[base + 2];

  float jn = sqrtf(jx * jx + jy * jy + jz * jz) + EPSF;
  float nx = jx / jn, ny = jy / jn, nz = jz / jn;

  float uz0 = -(nx * nx + ny * ny) / (nz + EPSF);
  float un  = sqrtf(nx * nx + ny * ny + uz0 * uz0) + EPSF;
  float ux = nx / un, uy = ny / un, uz = uz0 / un;

  float vx = uy * nz - uz * ny;              // V = U x nJ
  float vy = uz * nx - ux * nz;
  float vz = ux * ny - uy * nx;

  float r0 = ux * x0 + vx * x1 + nx * x2;    // RtX
  float r1 = uy * x0 + vy * x1 + ny * x2;
  float r2 = uz * x0 + vz * x1 + nz * x2;

  float av[3], bv[3], cv[3];
  av[0] = ux * r0 + uy * r1;  bv[0] = uy * r0 - ux * r1;  cv[0] = uz * r2;
  av[1] = vx * r0 + vy * r1;  bv[1] = vy * r0 - vx * r1;  cv[1] = vz * r2;
  av[2] = nx * r0 + ny * r1;  bv[2] = ny * r0 - nx * r1;  cv[2] = nz * r2;

#pragma unroll
  for (int i = 0; i < 3; ++i) {
    size_t cb = ((size_t)i * CPI + bn) * K1;
    T[cb + 0 * EQ + e] = (__bf16)av[i];
    T[cb + 1 * EQ + e] = (__bf16)bv[i];
    T[cb + 2 * EQ + e] = (__bf16)cv[i];
  }
}

// ---------------------------------------------------------------------------
// Shared WMMA fragment helpers (ISA 16-bit A 16x32 / B 32x16 layouts)
// ---------------------------------------------------------------------------
__device__ __forceinline__ void load_a_frags(const u32x4* Apan, int wm, int l,
                                             int half, Frag16 af[2]) {
#pragma unroll
  for (int mt = 0; mt < 2; ++mt) {
    int row = wm * 32 + mt * 16 + (l & 15);
    int idx = row * 4 + half;                // lanes<16: K 0..7, >=16: K 8..15
    af[mt].q[0] = Apan[idx];
    af[mt].q[1] = Apan[idx + 2];             // K +16
  }
}
__device__ __forceinline__ void load_b_frags(const u32x4* Bpan, int wn, int l,
                                             int half, Frag16 bf[2]) {
#pragma unroll
  for (int nt = 0; nt < 2; ++nt) {
    int col = wn * 32 + nt * 16 + (l & 15);
    int idx = col * 4 + half;                // lanes<16: K 0..7, >=16: K 8..15
    bf[nt].q[0] = Bpan[idx];
    bf[nt].q[1] = Bpan[idx + 2];             // K +16
  }
}
__device__ __forceinline__ void wmma_2x2(const Frag16 af[2], const Frag16 bf[2],
                                         v8f acc[2][2]) {
#pragma unroll
  for (int mt = 0; mt < 2; ++mt)
#pragma unroll
    for (int nt = 0; nt < 2; ++nt)
      acc[mt][nt] = __builtin_amdgcn_wmma_f32_16x16x32_bf16(
          false, af[mt].v, false, bf[nt].v, (short)0, acc[mt][nt],
          false, false);
}
// Re-stage a wave's 32x32 f32 tile in LDS, then store rows lane-contiguously.
__device__ __forceinline__ void store_tile(float* S, v8f acc[2][2], int l,
                                           int half, float* dst, int rbase,
                                           int cbase2, int ldd) {
#pragma unroll
  for (int mt = 0; mt < 2; ++mt)
#pragma unroll
    for (int nt = 0; nt < 2; ++nt) {
      int col32 = nt * 16 + (l & 15);
#pragma unroll
      for (int r = 0; r < 8; ++r)
        S[col32 * 33 + (mt * 16 + r + half * 8)] = acc[mt][nt][r];
    }
  for (int row32 = 0; row32 < 32; ++row32)
    dst[(size_t)(rbase + row32) * ldd + cbase2 + l] = S[l * 33 + row32];
}

// ---------------------------------------------------------------------------
// GEMM1: Y[f][col] = sum_k Wcat[f][k] * T[col][k]
// 128x64 block tile, 8 waves (4x2), wave = 2x2 wmma tiles.
// TDM path: double-buffered LDS panels filled by tensor_load_to_lds, one
// barrier per K-step, DMA of step k+1 overlapped with WMMA of step k.
// ---------------------------------------------------------------------------
__global__ __launch_bounds__(256) void gemm1_kernel(
    const __bf16* __restrict__ Wc, const __bf16* __restrict__ T,
    float* __restrict__ Y) {
  __shared__ __align__(16) u32x4 Alds[2][512];       // 2 x (128 rows x 32 k)
  __shared__ __align__(16) u32x4 Blds[2][256];       // 2 x (64 cols x 32 k)
  __shared__ float Stage[8 * 32 * 33];

  const int t   = threadIdx.x;
  const int l   = t & 31;
  const int wid = t >> 5;
  const int wm  = wid >> 1;
  const int wn  = wid & 1;
  const int mbase = blockIdx.x * 128;
  const int cbase = blockIdx.y * 64;
  const int half  = (l < 16) ? 0 : 1;

  v8f acc[2][2] = {};

#if HAVE_TDM
  const unsigned aOff[2] = {(unsigned)(uintptr_t)&Alds[0][0],
                            (unsigned)(uintptr_t)&Alds[1][0]};
  const unsigned bOff[2] = {(unsigned)(uintptr_t)&Blds[0][0],
                            (unsigned)(uintptr_t)&Blds[1][0]};
  if (wid == 0) {                                    // prologue DMA, buffer 0
    tdm_load_2d(aOff[0], Wc + (size_t)mbase * K1, 32, 128, K1, K1);
    tdm_load_2d(bOff[0], T + (size_t)cbase * K1, 32, 64, K1, K1);
  }
  for (int it = 0; it < K1 / 32; ++it) {
    const int cur = it & 1;
    if (wid == 0) __builtin_amdgcn_s_wait_tensorcnt(0);
    __syncthreads();                                 // buffer `cur` ready
    if (wid == 0 && it + 1 < K1 / 32) {              // DMA next while computing
      int kn = (it + 1) * 32;
      tdm_load_2d(aOff[cur ^ 1], Wc + (size_t)mbase * K1 + kn, 32, 128, K1, K1);
      tdm_load_2d(bOff[cur ^ 1], T + (size_t)cbase * K1 + kn, 32, 64, K1, K1);
    }
    Frag16 af[2], bf[2];
    load_a_frags(&Alds[cur][0], wm, l, half, af);
    load_b_frags(&Blds[cur][0], wn, l, half, bf);
    wmma_2x2(af, bf, acc);
  }
#else
  const u32x4* asrc = reinterpret_cast<const u32x4*>(Wc);
  const u32x4* bsrc = reinterpret_cast<const u32x4*>(T);
  for (int kk = 0; kk < K1; kk += 32) {
    int c0 = t, c1 = t + 256;
    Alds[0][c0] = asrc[((size_t)(mbase + (c0 >> 2)) * K1 + kk) / 8 + (c0 & 3)];
    Alds[0][c1 - 512 + 512] =
        asrc[((size_t)(mbase + (c1 >> 2)) * K1 + kk) / 8 + (c1 & 3)];
    Blds[0][t] = bsrc[((size_t)(cbase + (t >> 2)) * K1 + kk) / 8 + (t & 3)];
    if (kk + 32 < K1)
      __builtin_prefetch(&T[(size_t)(cbase + (t >> 2)) * K1 + kk + 32], 0, 1);
    __syncthreads();
    Frag16 af[2], bf[2];
    load_a_frags(&Alds[0][0], wm, l, half, af);
    load_b_frags(&Blds[0][0], wn, l, half, bf);
    wmma_2x2(af, bf, acc);
    __syncthreads();
  }
#endif
  store_tile(&Stage[wid * (32 * 33)], acc, l, half, Y, mbase + wm * 32,
             cbase + wn * 32, NCOL);
}

// ---------------------------------------------------------------------------
// GEMM2: D[g][col] = sum_f Wb[g][f] * Y[f][col]
// B panel is f32 Y -> bf16 convert + transpose in VGPRs (so no TDM here).
// ---------------------------------------------------------------------------
__global__ __launch_bounds__(256) void gemm2_kernel(
    const __bf16* __restrict__ Wb, const float* __restrict__ Yf,
    float* __restrict__ D) {
  __shared__ __align__(16) u32x4 Alds[512];
  __shared__ __align__(16) __bf16 BldsH[64 * 32];
  __shared__ float Stage[8 * 32 * 33];

  const int t   = threadIdx.x;
  const int l   = t & 31;
  const int wid = t >> 5;
  const int wm  = wid >> 1;
  const int wn  = wid & 1;
  const int mbase = blockIdx.x * 128;
  const int cbase = blockIdx.y * 64;
  const int half  = (l < 16) ? 0 : 1;

  const u32x4* asrc = reinterpret_cast<const u32x4*>(Wb);
  const u32x4* blds = reinterpret_cast<const u32x4*>(BldsH);

  v8f acc[2][2] = {};

  for (int kk = 0; kk < K2; kk += 32) {
    {
      int c0 = t, c1 = t + 256;
      Alds[c0] = asrc[((size_t)(mbase + (c0 >> 2)) * K2 + kk) / 8 + (c0 & 3)];
      Alds[c1] = asrc[((size_t)(mbase + (c1 >> 2)) * K2 + kk) / 8 + (c1 & 3)];
      int fl = t >> 3;                               // 0..31
      int cg = t & 7;                                // 8 cols per thread
      const float* p = Yf + (size_t)(kk + fl) * NCOL + cbase + cg * 8;
      f32x4 lo = *reinterpret_cast<const f32x4*>(p);
      f32x4 hi = *reinterpret_cast<const f32x4*>(p + 4);
      if (kk + 32 < K2) __builtin_prefetch(p + (size_t)32 * NCOL, 0, 1);
#pragma unroll
      for (int j = 0; j < 4; ++j) {
        BldsH[(cg * 8 + j) * 32 + fl]     = (__bf16)lo[j];
        BldsH[(cg * 8 + 4 + j) * 32 + fl] = (__bf16)hi[j];
      }
    }
    __syncthreads();
    Frag16 af[2], bf[2];
    load_a_frags(&Alds[0], wm, l, half, af);
    load_b_frags(&blds[0], wn, l, half, bf);
    wmma_2x2(af, bf, acc);
    __syncthreads();
  }
  store_tile(&Stage[wid * (32 * 33)], acc, l, half, D, mbase + wm * 32,
             cbase + wn * 32, NCOL);
}

// ---------------------------------------------------------------------------
// Vector-neuron leaky ReLU (col = i*CPI + b*NQ + n -> all streams coalesced)
// ---------------------------------------------------------------------------
__global__ __launch_bounds__(256) void vnlrelu_kernel(
    const float* __restrict__ Yf, const float* __restrict__ Df,
    float* __restrict__ out) {
  const int bi = blockIdx.x;                 // 4096
  const int f  = bi >> 4;
  const int b  = bi & 15;
  for (int c = 0; c < 4; ++c) {
    int n = threadIdx.x + c * 256;
    size_t base = (size_t)f * NCOL + (size_t)b * NQ + n;
    float x0 = Yf[base], x1 = Yf[base + CPI], x2 = Yf[base + 2 * CPI];
    float d0 = Df[base], d1 = Df[base + CPI], d2 = Df[base + 2 * CPI];
    float dot = x0 * d0 + x1 * d1 + x2 * d2;
    float dns = d0 * d0 + d1 * d1 + d2 * d2;
    float s   = dot / (dns + EPSF);
    bool  pos = (dot >= 0.0f);
    float l0 = pos ? x0 : x0 - s * d0;
    float l1 = pos ? x1 : x1 - s * d1;
    float l2 = pos ? x2 : x2 - s * d2;
    size_t ob = (((size_t)b * FQ + f) * 3) * NQ + n;
    out[ob + 0 * NQ] = 0.2f * x0 + 0.8f * l0;
    out[ob + 1 * NQ] = 0.2f * x1 + 0.8f * l1;
    out[ob + 2 * NQ] = 0.2f * x2 + 0.8f * l2;
  }
}

// ---------------------------------------------------------------------------
// Launch
// ---------------------------------------------------------------------------
extern "C" void kernel_launch(void* const* d_in, const int* in_sizes, int n_in,
                              void* d_out, int out_size, void* d_ws, size_t ws_size,
                              hipStream_t stream) {
  const float* X  = (const float*)d_in[0];
  const float* J  = (const float*)d_in[1];
  const float* A  = (const float*)d_in[2];
  const float* Bw = (const float*)d_in[3];
  const float* Cw = (const float*)d_in[4];
  const float* W  = (const float*)d_in[5];
  float* out = (float*)d_out;

  char* ws = (char*)d_ws;
  const size_t szT  = (size_t)NCOL * K1 * sizeof(__bf16);   // 75,497,472
  const size_t szY  = (size_t)FQ * NCOL * sizeof(float);    // 50,331,648
  const size_t szD  = szY;
  const size_t szWc = (size_t)FQ * K1 * sizeof(__bf16);     // 393,216

  __bf16* Tbf  = (__bf16*)(ws);
  float*  Yf   = (float*)(ws + szT);
  float*  Df   = (float*)(ws + szT + szY);
  __bf16* Wcat = (__bf16*)(ws + szT + szY + szD);
  __bf16* Wb   = (__bf16*)(ws + szT + szY + szD + szWc);

  wprep_kernel<<<dim3((FQ * K1 + FQ * FQ) / 256), dim3(256), 0, stream>>>(
      A, Bw, Cw, W, Wcat, Wb);
  prep_kernel<<<dim3(BQ * NQ), dim3(256), 0, stream>>>(X, J, Tbf);
  gemm1_kernel<<<dim3(FQ / 128, NCOL / 64), dim3(256), 0, stream>>>(Wcat, Tbf, Yf);
  gemm2_kernel<<<dim3(FQ / 128, NCOL / 64), dim3(256), 0, stream>>>(Wb, Yf, Df);
  vnlrelu_kernel<<<dim3(FQ * BQ), dim3(256), 0, stream>>>(Yf, Df, out);
}
